// MultiHeadAttention_31765578121852
// MI455X (gfx1250) — matrix operand order repaired and checked by reading, not compiled, not run on hardware
//
#include <hip/hip_runtime.h>

// ---------------------------------------------------------------------------
// MHA forward for B=2, T=2048, d_model=2048, 16 heads x 128.
// bf16 WMMA (v_wmma_f32_16x16x32_bf16) everywhere; f32 accumulation.
// Attention K tiles staged via global_load_async_to_lds_b128 (ASYNCcnt).
// ws layout (bf16 elements):
//   [0,        8388608)  q  (B*H, T, 128)   roped in-place by k_rope
//   [8388608, 16777216)  k  (B*H, T, 128)   roped in-place by k_rope
//   [16777216,25165824)  v  (B*H, T, 128)
//   [25165824,33554432)  attn out (B, T, 2048)
// d_out (float): out[8388608] | k[8388608] | v[8388608]
// ---------------------------------------------------------------------------

typedef __bf16 bhalf;
typedef __attribute__((ext_vector_type(16))) __bf16 v16bf;
typedef __attribute__((ext_vector_type(8)))  float  v8f;

#define BQ  2
#define TQ  2048
#define NH  16
#define HD  128
#define DM  2048
#define SEC 8388608   // elements per q/k/v section

union FragU { v16bf v; bhalf e[16]; uint4 q[2]; };

__device__ inline v8f wmma_bf16(v16bf a, v16bf b, v8f c) {
  return __builtin_amdgcn_wmma_f32_16x16x32_bf16(false, a, false, b,
                                                 (short)0, c, false, false);
}

// A fragment (16x32, M x K), row-major bf16 source, K contiguous.
// lane L: M = L%16 ; e[i] holds K = 16*(i/8) + 8*(L/16) + i%8
__device__ inline v16bf load_a_frag(const bhalf* base, int ld, int lane) {
  int row = lane & 15;
  int kh  = (lane >> 4) * 8;
  const bhalf* p = base + row * ld + kh;
  FragU f;
  f.q[0] = *(const uint4*)(p);
  f.q[1] = *(const uint4*)(p + 16);
  return f.v;
}

// A fragment from fp32 source with on-the-fly bf16 conversion.
__device__ inline v16bf load_a_frag_f32(const float* base, int ld, int lane) {
  int row = lane & 15;
  int kh  = (lane >> 4) * 8;
  const float* p = base + row * ld + kh;
  float4 a0 = *(const float4*)(p);
  float4 a1 = *(const float4*)(p + 4);
  float4 a2 = *(const float4*)(p + 16);
  float4 a3 = *(const float4*)(p + 20);
  FragU f;
  f.e[0]=(bhalf)a0.x; f.e[1]=(bhalf)a0.y; f.e[2]=(bhalf)a0.z; f.e[3]=(bhalf)a0.w;
  f.e[4]=(bhalf)a1.x; f.e[5]=(bhalf)a1.y; f.e[6]=(bhalf)a1.z; f.e[7]=(bhalf)a1.w;
  f.e[8]=(bhalf)a2.x; f.e[9]=(bhalf)a2.y; f.e[10]=(bhalf)a2.z; f.e[11]=(bhalf)a2.w;
  f.e[12]=(bhalf)a3.x; f.e[13]=(bhalf)a3.y; f.e[14]=(bhalf)a3.z; f.e[15]=(bhalf)a3.w;
  return f.v;
}

// B fragment (32x16, K x N) from a [N][K] (K contiguous) view, stride ld.
// lane L: N = L%16 ; e[i] holds K = 16*(i/8) + 8*(L/16) + i%8
__device__ inline v16bf load_b_frag(const bhalf* base, int ld, int lane) {
  int col = lane & 15;
  int kh  = (lane >> 4) * 8;
  const bhalf* p = base + col * ld + kh;
  FragU f;
  f.q[0] = *(const uint4*)(p);
  f.q[1] = *(const uint4*)(p + 16);
  return f.v;
}

// Async global -> LDS copy of 16 bytes (GVS mode; tracked by ASYNCcnt).
__device__ inline void async_copy_b128(const void* sbase, unsigned voff_bytes,
                                       void* lds_ptr) {
  unsigned lds_addr = (unsigned)(size_t)lds_ptr;   // low 32 bits = LDS offset
  asm volatile("global_load_async_to_lds_b128 %0, %1, %2"
               :: "v"(lds_addr), "v"(voff_bytes), "s"(sbase)
               : "memory");
}
__device__ inline void wait_asynccnt0() {
  asm volatile("s_wait_asynccnt 0x0" ::: "memory");
}

// ---------------------------------------------------------------------------
// Kernel 1: qkv = x @ Wqkv  (M=4096, K=2048, N=6144), scatter bf16 q/k/v.
// Block tile 128x128, 8 waves as 4(M) x 2(N); each wave 32x64.
// ---------------------------------------------------------------------------
__global__ __launch_bounds__(256) void k_gemm_qkv(const float* __restrict__ x,
                                                  const float* __restrict__ w,
                                                  bhalf* __restrict__ ws) {
  __shared__ __align__(16) bhalf Bs[128 * 32];   // [n][k] transposed
  const int tid  = threadIdx.x;
  const int lane = tid & 31, wid = tid >> 5;
  const int n0 = blockIdx.x * 128;
  const int m0 = blockIdx.y * 128;
  const int wm = (wid >> 1) * 32;
  const int wn = (wid & 1) * 64;

  v8f acc[2][4];
#pragma unroll
  for (int i = 0; i < 2; i++)
#pragma unroll
    for (int j = 0; j < 4; j++) acc[i][j] = (v8f){0.f,0.f,0.f,0.f,0.f,0.f,0.f,0.f};

  for (int k0 = 0; k0 < DM; k0 += 32) {
    __syncthreads();
#pragma unroll
    for (int i = 0; i < 4; i++) {
      int idx = tid + i * 256;            // 0..1023
      int row = idx >> 5;                 // k  0..31
      int c4  = idx & 31;                 // n/4
      float4 wv = *(const float4*)(w + (size_t)(k0 + row) * (3 * DM) + n0 + c4 * 4);
      if (k0 + 32 < DM)                   // prefetch next weight tile
        __builtin_prefetch(w + (size_t)(k0 + 32 + row) * (3 * DM) + n0 + c4 * 4, 0, 3);
      int nn = c4 * 4;
      Bs[(nn + 0) * 32 + row] = (bhalf)wv.x;
      Bs[(nn + 1) * 32 + row] = (bhalf)wv.y;
      Bs[(nn + 2) * 32 + row] = (bhalf)wv.z;
      Bs[(nn + 3) * 32 + row] = (bhalf)wv.w;
    }
    __syncthreads();

    v16bf af0 = load_a_frag_f32(x + (size_t)(m0 + wm) * DM + k0, DM, lane);
    v16bf af1 = load_a_frag_f32(x + (size_t)(m0 + wm + 16) * DM + k0, DM, lane);
#pragma unroll
    for (int nt = 0; nt < 4; nt++) {
      v16bf bf = load_b_frag(&Bs[(wn + nt * 16) * 32], 32, lane);
      acc[0][nt] = wmma_bf16(af0, bf, acc[0][nt]);
      acc[1][nt] = wmma_bf16(af1, bf, acc[1][nt]);
    }
  }

  const int rowl = (lane >> 4) << 3;
#pragma unroll
  for (int mt = 0; mt < 2; mt++) {
#pragma unroll
    for (int nt = 0; nt < 4; nt++) {
#pragma unroll
      for (int r = 0; r < 8; r++) {
        int m = m0 + wm + mt * 16 + r + rowl;
        int n = n0 + wn + nt * 16 + (lane & 15);
        int b = m >> 11, t = m & (TQ - 1);
        int which = n >> 11;
        int rem = n & (DM - 1);
        int h = rem >> 7, d = rem & (HD - 1);
        bhalf* dst = ws + (size_t)which * SEC;
        dst[(((size_t)(b * NH + h) * TQ + t) * HD) + d] = (bhalf)acc[mt][nt][r];
      }
    }
  }
}

// ---------------------------------------------------------------------------
// Kernel 2: RoPE on q,k (in place, bf16) + export fp32 k (roped) and v.
// One thread per (bh, t, pair).  4,194,304 threads.
// ---------------------------------------------------------------------------
__global__ __launch_bounds__(256) void k_rope(bhalf* __restrict__ ws,
                                              float* __restrict__ out) {
  int gi = blockIdx.x * 256 + threadIdx.x;
  int p  = gi & 63;
  int t  = (gi >> 6) & (TQ - 1);
  int bh = gi >> 17;
  int pm = p & 31;
  float fc = powf(10000.0f, -(float)(2 * pm)     * (1.0f / 64.0f));
  float fs = powf(10000.0f, -(float)(2 * pm + 1) * (1.0f / 64.0f));
  float tf = (float)t;
  float c = cosf(tf * fc);
  float s = sinf(tf * fs);
  size_t base = ((size_t)bh * TQ + t) * HD + 2 * p;
  { // q
    bhalf* q = ws + base;
    float x1 = (float)q[0], x2 = (float)q[1];
    q[0] = (bhalf)(x1 * c - x2 * s);
    q[1] = (bhalf)(x1 * s + x2 * c);
  }
  { // k (roped) -> ws bf16 + d_out fp32
    bhalf* k = ws + SEC + base;
    float x1 = (float)k[0], x2 = (float)k[1];
    float e = x1 * c - x2 * s, o = x1 * s + x2 * c;
    k[0] = (bhalf)e; k[1] = (bhalf)o;
    float* ko = out + SEC + base;
    ko[0] = e; ko[1] = o;
  }
  { // v copy -> d_out fp32
    bhalf* v = ws + 2 * (size_t)SEC + base;
    float* vo = out + 2 * (size_t)SEC + base;
    vo[0] = (float)v[0];
    vo[1] = (float)v[1];
  }
}

// ---------------------------------------------------------------------------
// Kernel 3: causal flash attention. Block = (bh, qtile of 128), 8 waves x 16 q.
// K tile staged once per block via async global->LDS DMA; V tile staged
// transposed; P round-trips per-wave LDS to convert C-layout -> A-layout.
// ---------------------------------------------------------------------------
__global__ __launch_bounds__(256) void k_attn(const bhalf* __restrict__ ws,
                                              bhalf* __restrict__ aws) {
  __shared__ __align__(16) bhalf Ks[32 * HD];        // [key][d] row-major
  __shared__ __align__(16) bhalf Vt[HD * 32];        // [d][key] transposed
  __shared__ __align__(16) bhalf Pt[8][16 * 32];     // per-wave P [row][key]
  const int lane = threadIdx.x & 31, wid = threadIdx.x >> 5;
  const int bh = blockIdx.x;
  const int q0 = blockIdx.y * 128;
  const int qr = q0 + wid * 16;

  const bhalf* qb = ws + ((size_t)bh * TQ) * HD;
  const bhalf* kb = ws + (size_t)SEC + ((size_t)bh * TQ) * HD;
  const bhalf* vb = ws + 2 * (size_t)SEC + ((size_t)bh * TQ) * HD;

  v16bf qf[4];
#pragma unroll
  for (int d = 0; d < 4; d++)
    qf[d] = load_a_frag(qb + (size_t)qr * HD + d * 32, HD, lane);

  v8f o[8];
#pragma unroll
  for (int i = 0; i < 8; i++) o[i] = (v8f){0.f,0.f,0.f,0.f,0.f,0.f,0.f,0.f};
  float mrow[8], lrow[8];
#pragma unroll
  for (int r = 0; r < 8; r++) { mrow[r] = -1e30f; lrow[r] = 0.f; }

  const int rowl = (lane >> 4) << 3;
  const float scale = 0.08838834764831845f;   // 1/sqrt(128)
  const int kend = q0 + 128;

  for (int kt = 0; kt < kend; kt += 32) {
    __syncthreads();          // previous tile fully consumed
    // --- async stage K tile (32 x 128 bf16 = 8KB), no transpose needed ---
#pragma unroll
    for (int i = 0; i < 2; i++) {
      int li  = threadIdx.x + i * 256;          // 0..511
      int key = li >> 4;                        // 0..31
      int dc  = (li & 15) * 8;                  // d chunk (8 bf16 = 16B)
      unsigned voff = (unsigned)(((kt + key) * HD + dc) * 2);
      async_copy_b128(kb, voff, &Ks[key * HD + dc]);
    }
    // --- stage V tile transposed (manual: needs transpose) ---
#pragma unroll
    for (int i = 0; i < 2; i++) {
      int li = threadIdx.x + i * 256;           // 0..511
      int key = li >> 4;                        // 0..31
      int dc  = (li & 15) * 8;
      union { uint4 u; bhalf e[8]; } uu;
      uu.u = *(const uint4*)(vb + (size_t)(kt + key) * HD + dc);
#pragma unroll
      for (int j = 0; j < 8; j++) Vt[(dc + j) * 32 + key] = uu.e[j];
    }
    wait_asynccnt0();
    __syncthreads();

    v8f sl = (v8f){0.f,0.f,0.f,0.f,0.f,0.f,0.f,0.f};
    v8f sr = sl;
#pragma unroll
    for (int d = 0; d < 4; d++) {
      v16bf kl = load_b_frag(&Ks[d * 32], HD, lane);            // keys 0..15
      v16bf kr = load_b_frag(&Ks[16 * HD + d * 32], HD, lane);  // keys 16..31
      sl = wmma_bf16(qf[d], kl, sl);
      sr = wmma_bf16(qf[d], kr, sr);
    }

    bhalf* pw = &Pt[wid][0];
    float alpha[8];
#pragma unroll
    for (int r = 0; r < 8; r++) {
      int q = qr + r + rowl;
      int keyl = kt + (lane & 15);
      float a  = sl[r] * scale; if (keyl > q)      a  = -1e30f;
      float b2 = sr[r] * scale; if (keyl + 16 > q) b2 = -1e30f;
      float tm = fmaxf(a, b2);
#pragma unroll
      for (int off = 8; off >= 1; off >>= 1)
        tm = fmaxf(tm, __shfl_xor(tm, off, 32));
      float mn = fmaxf(mrow[r], tm);
      float pa = __expf(a - mn);
      float pb = __expf(b2 - mn);
      alpha[r] = __expf(mrow[r] - mn);
      float rs = pa + pb;
#pragma unroll
      for (int off = 8; off >= 1; off >>= 1)
        rs += __shfl_xor(rs, off, 32);
      lrow[r] = lrow[r] * alpha[r] + rs;
      mrow[r] = mn;
      int prow = r + rowl;
      pw[prow * 32 + (lane & 15)]      = (bhalf)pa;
      pw[prow * 32 + 16 + (lane & 15)] = (bhalf)pb;
    }
#pragma unroll
    for (int dt = 0; dt < 8; dt++)
#pragma unroll
      for (int r = 0; r < 8; r++) o[dt][r] *= alpha[r];

    v16bf pf = load_a_frag(pw, 32, lane);       // per-wave LDS, in-order DS
#pragma unroll
    for (int dt = 0; dt < 8; dt++) {
      v16bf vf = load_b_frag(&Vt[dt * 16 * 32], 32, lane);
      o[dt] = wmma_bf16(pf, vf, o[dt]);
    }
  }

  const int b = bh >> 4, h = bh & 15;
#pragma unroll
  for (int dt = 0; dt < 8; dt++) {
#pragma unroll
    for (int r = 0; r < 8; r++) {
      int q = qr + r + rowl;
      int d = dt * 16 + (lane & 15);
      float val = o[dt][r] / lrow[r];
      aws[((size_t)(b * TQ + q)) * DM + h * HD + d] = (bhalf)val;
    }
  }
}

// ---------------------------------------------------------------------------
// Kernel 4: out = attn (bf16, M=4096,K=2048) @ Wout (fp32, 2048x2048) -> fp32
// ---------------------------------------------------------------------------
__global__ __launch_bounds__(256) void k_gemm_out(const bhalf* __restrict__ a,
                                                  const float* __restrict__ w,
                                                  float* __restrict__ out) {
  __shared__ __align__(16) bhalf Bs[128 * 32];
  const int tid  = threadIdx.x;
  const int lane = tid & 31, wid = tid >> 5;
  const int n0 = blockIdx.x * 128;
  const int m0 = blockIdx.y * 128;
  const int wm = (wid >> 1) * 32;
  const int wn = (wid & 1) * 64;

  v8f acc[2][4];
#pragma unroll
  for (int i = 0; i < 2; i++)
#pragma unroll
    for (int j = 0; j < 4; j++) acc[i][j] = (v8f){0.f,0.f,0.f,0.f,0.f,0.f,0.f,0.f};

  for (int k0 = 0; k0 < DM; k0 += 32) {
    __syncthreads();
#pragma unroll
    for (int i = 0; i < 4; i++) {
      int idx = tid + i * 256;
      int row = idx >> 5;
      int c4  = idx & 31;
      float4 wv = *(const float4*)(w + (size_t)(k0 + row) * DM + n0 + c4 * 4);
      if (k0 + 32 < DM)
        __builtin_prefetch(w + (size_t)(k0 + 32 + row) * DM + n0 + c4 * 4, 0, 3);
      int nn = c4 * 4;
      Bs[(nn + 0) * 32 + row] = (bhalf)wv.x;
      Bs[(nn + 1) * 32 + row] = (bhalf)wv.y;
      Bs[(nn + 2) * 32 + row] = (bhalf)wv.z;
      Bs[(nn + 3) * 32 + row] = (bhalf)wv.w;
    }
    __syncthreads();

    v16bf af0 = load_a_frag(a + (size_t)(m0 + wm) * DM + k0, DM, lane);
    v16bf af1 = load_a_frag(a + (size_t)(m0 + wm + 16) * DM + k0, DM, lane);
#pragma unroll
    for (int nt = 0; nt < 4; nt++) {
      v16bf bf = load_b_frag(&Bs[(wn + nt * 16) * 32], 32, lane);
      acc[0][nt] = wmma_bf16(af0, bf, acc[0][nt]);
      acc[1][nt] = wmma_bf16(af1, bf, acc[1][nt]);
    }
  }

  const int rowl = (lane >> 4) << 3;
#pragma unroll
  for (int mt = 0; mt < 2; mt++) {
#pragma unroll
    for (int nt = 0; nt < 4; nt++) {
#pragma unroll
      for (int r = 0; r < 8; r++) {
        int m = m0 + wm + mt * 16 + r + rowl;
        int n = n0 + wn + nt * 16 + (lane & 15);
        out[(size_t)m * DM + n] = acc[mt][nt][r];
      }
    }
  }
}

// ---------------------------------------------------------------------------
extern "C" void kernel_launch(void* const* d_in, const int* in_sizes, int n_in,
                              void* d_out, int out_size, void* d_ws, size_t ws_size,
                              hipStream_t stream) {
  const float* x    = (const float*)d_in[0];
  // d_in[1] = attn_mask (causal tril) — hardcoded in k_attn
  const float* Wqkv = (const float*)d_in[2];
  const float* Wout = (const float*)d_in[3];
  float* out = (float*)d_out;
  bhalf* ws  = (bhalf*)d_ws;
  bhalf* aws = ws + 3 * (size_t)SEC;   // attn output, bf16

  k_gemm_qkv<<<dim3(48, 32), 256, 0, stream>>>(x, Wqkv, ws);
  k_rope<<<dim3((BQ * NH * TQ * 64) / 256), 256, 0, stream>>>(ws, out);
  k_attn<<<dim3(BQ * NH, TQ / 128), 256, 0, stream>>>(ws, aws);
  k_gemm_out<<<dim3(16, 32), 256, 0, stream>>>(aws, Wout, out);
}
